// LGMformer_37022618091526
// MI455X (gfx1250) — matrix operand order repaired and checked by reading, not compiled, not run on hardware
//
#include <hip/hip_runtime.h>
#include <hip/hip_bf16.h>
#include <math.h>

#define DEVINL __device__ __forceinline__

typedef __attribute__((ext_vector_type(16))) _Float16 v16h;
typedef __attribute__((ext_vector_type(8)))  _Float16 v8h;
typedef __attribute__((ext_vector_type(8)))  float    v8f;
typedef __attribute__((ext_vector_type(8)))  float    v8fv;
typedef __attribute__((ext_vector_type(4)))  float    v4f;

constexpr int   Nn     = 50000;
constexpr int   Ee     = 400000;
constexpr int   Hh     = 4;
constexpr int   Kk     = 128;
constexpr int   Ss     = 3;          // HOPS+1
constexpr float SCL    = 0.25f;      // D^-0.5, D=16
constexpr float LN_EPS = 1e-5f;

// ---------------- WMMA helpers (gfx1250: v_wmma_f32_16x16x32_f16) ----------------

DEVINL v8f wmma16(v16h a, v16h b, v8f c) {
  return __builtin_amdgcn_wmma_f32_16x16x32_f16(false, a, false, b, (short)0, c,
                                                false, false);
}

// A fragment: 16x32 tile of a row-major [*,64] matrix (lda=64).
// ISA layout: M=lane&15; k = e + 8*hi (+8 for e>=8) -> two contiguous 8-half runs.
DEVINL v16h loadA(const _Float16* base, int lane) {
  const int m = lane & 15, hi = lane >> 4;
  const _Float16* p = base + m * 64 + 8 * hi;
  v8h x0 = *(const v8h*)p;            // 16B aligned
  v8h x1 = *(const v8h*)(p + 16);
  return __builtin_shufflevector(x0, x1, 0, 1, 2, 3, 4, 5, 6, 7,
                                 8, 9, 10, 11, 12, 13, 14, 15);
}

DEVINL v16h loadA(const float* base, int lane) {
  const int m = lane & 15, hi = lane >> 4;
  const float* p = base + m * 64 + 8 * hi;
  v8fv x0 = *(const v8fv*)p;          // 32B aligned
  v8fv x1 = *(const v8fv*)(p + 16);
  v16h a;
#pragma unroll
  for (int e = 0; e < 8; ++e) {
    a[e]     = (_Float16)x0[e];
    a[e + 8] = (_Float16)x1[e];
  }
  return a;
}

// B fragment from PRE-SWIZZLED weight: layout [ct][kc][lane][16 halfs] contiguous.
// (packed so that b[e] = W[((e&7) + 8*hi + 16*(e>>3) + 32*kc)*64 + ct*16 + n], n=lane&15)
DEVINL v16h loadBpk(const _Float16* Wpk, int ct, int kc, int lane) {
  return *(const v16h*)(Wpk + (((ct * 2 + kc) * 32 + lane) << 4));  // 32B aligned
}

// 16 contiguous f16 -> float[16] via two b128 loads
DEVINL void ld16(const _Float16* p, float* out) {
  v8h x0 = *(const v8h*)p;
  v8h x1 = *(const v8h*)(p + 8);
#pragma unroll
  for (int d = 0; d < 8; ++d) {
    out[d]     = (float)x0[d];
    out[d + 8] = (float)x1[d];
  }
}

// ---------------- generic 64-wide GEMM: out[M,64] = A[M,64] @ W[64,64] + bias ----
// MODE: 0=f32 out, 1=f16 out, 2=f16 out + exact GELU,
//       3=f32 out + residual, 4=f32 out, (resid+v)*hw0 + og*hw1  (final combine)
// MT row-tiles per block; B fragments staged in LDS via async DMA and reused.
template <typename AT, int MODE, int MT>
__global__ void __launch_bounds__(128)
k_gemm64(const AT* __restrict__ A, const _Float16* __restrict__ W,
         const float* __restrict__ bias, void* __restrict__ out,
         const float* __restrict__ resid, const float* __restrict__ og,
         const float* __restrict__ hw) {
  __shared__ _Float16 wlds[4096];     // packed 64x64 f16 weight, 8KB

  // stage weight into LDS with gfx1250 async copies (ASYNCcnt-tracked)
  {
    const int t = threadIdx.x;        // 128 threads x 16B x 4 = 8KB
#pragma unroll
    for (int i = 0; i < 4; ++i) {
      const char* gp = (const char*)W + t * 16 + i * 2048;
      unsigned    lp = (unsigned)(size_t)(&wlds[0]) + (unsigned)(t * 16 + i * 2048);
      asm volatile("global_load_async_to_lds_b128 %0, %1, off"
                   :: "v"(lp), "v"(gp) : "memory");
    }
    asm volatile("s_wait_asynccnt 0x0" ::: "memory");
    __syncthreads();
  }

  const int lane = threadIdx.x & 31;
  const int wv   = threadIdx.x >> 5;  // 0..3 -> 16-col tile
  const int m    = lane & 15, hi = lane >> 4;
  const int col  = wv * 16 + m;
  const float bn = bias[col];
  float h0 = 0.f, h1 = 0.f;
  if constexpr (MODE == 4) { h0 = hw[0]; h1 = hw[1]; }

  v16h b0 = loadBpk(wlds, wv, 0, lane);   // held across all MT row tiles
  v16h b1 = loadBpk(wlds, wv, 1, lane);

#pragma unroll
  for (int mt = 0; mt < MT; ++mt) {
    const size_t mrow = ((size_t)blockIdx.x * MT + mt) * 16;
    v16h a0 = loadA(A + mrow * 64, lane);
    v16h a1 = loadA(A + mrow * 64 + 32, lane);
    v8f c = {};
    c = wmma16(a0, b0, c);
    c = wmma16(a1, b1, c);

#pragma unroll
    for (int r = 0; r < 8; ++r) {
      size_t row = mrow + (size_t)(r + 8 * hi);
      size_t idx = row * 64 + col;
      float v = c[r] + bn;
      if constexpr (MODE == 0) {
        ((float*)out)[idx] = v;
      } else if constexpr (MODE == 1) {
        ((_Float16*)out)[idx] = (_Float16)v;
      } else if constexpr (MODE == 2) {
        float g = 0.5f * v * (1.0f + erff(v * 0.70710678118f));
        ((_Float16*)out)[idx] = (_Float16)g;
      } else if constexpr (MODE == 3) {
        ((float*)out)[idx] = v + resid[idx];
      } else {
        ((float*)out)[idx] = (resid[idx] + v) * h0 + og[idx] * h1;
      }
    }
  }
}

// ---------------- fused NTIformer edge block --------------------------------------
// 32 edges / workgroup (128 token rows). gather -> LN -> QKV WMMA -> 1x4 attn
// -> O-proj WMMA -> residual -> atomic scatter-add into out_cen accumulator.
// Each wave owns one 16-column tile (ct == wave id): B fragments and biases are
// loaded once and held in registers; K and V share each A fragment.
__global__ void __launch_bounds__(128)
k_edge(const float* __restrict__ h, const int* __restrict__ srcI,
       const int* __restrict__ dstI, const float* __restrict__ lnS,
       const float* __restrict__ lnB, const _Float16* __restrict__ Wq,
       const float* __restrict__ bq, const _Float16* __restrict__ Wk,
       const float* __restrict__ bk, const _Float16* __restrict__ Wv,
       const float* __restrict__ bv, const _Float16* __restrict__ Wo,
       const float* __restrict__ bo, float* __restrict__ acc) {
  __shared__ _Float16 tok[128 * 64];   // LN'd tokens (4 per edge)
  __shared__ _Float16 khs[128 * 64];
  __shared__ _Float16 vhs[128 * 64];
  __shared__ _Float16 qin[32 * 64];    // compacted token-0 rows
  __shared__ _Float16 qhs[32 * 64];
  __shared__ _Float16 ohs[32 * 64];    // attention output

  const int tid  = threadIdx.x;
  const int lane = tid & 31;
  const int wv   = tid >> 5;
  const int eb   = blockIdx.x * 32;
  const int m    = lane & 15, hi = lane >> 4;
  const int col  = wv * 16 + m;

  // ---- build tokens + LayerNorm (one token row per thread, two-pass) ----
  {
    const int el = tid >> 2, t = tid & 3;
    const int ge = eb + el;
    const float* hc = h + (size_t)dstI[ge] * 64;   // x_cen
    const float* hn = h + (size_t)srcI[ge] * 64;   // x_nei
    float sum = 0.f, sq = 0.f;
#pragma unroll
    for (int c = 0; c < 64; ++c) {
      float xc = hc[c], xn = hn[c];
      float tv = (t == 0) ? xc : (t == 1) ? xn : (t == 2) ? (xn + xc) : (xn - xc);
      sum += tv; sq += tv * tv;
    }
    float mean = sum * (1.f / 64.f);
    float var  = sq * (1.f / 64.f) - mean * mean;
    float rstd = rsqrtf(var + LN_EPS);
#pragma unroll
    for (int c = 0; c < 64; ++c) {
      float xc = hc[c], xn = hn[c];
      float tv = (t == 0) ? xc : (t == 1) ? xn : (t == 2) ? (xn + xc) : (xn - xc);
      tok[tid * 64 + c] = (_Float16)(((tv - mean) * rstd) * lnS[c] + lnB[c]);
    }
  }
  __syncthreads();
  for (int i = tid; i < 32 * 64; i += 128)
    qin[i] = tok[(i >> 6) * 4 * 64 + (i & 63)];
  __syncthreads();

  // ---- K + V projections (shared A fragments), then Q ----
  {
    v16h kb0 = loadBpk(Wk, wv, 0, lane), kb1 = loadBpk(Wk, wv, 1, lane);
    v16h vb0 = loadBpk(Wv, wv, 0, lane), vb1 = loadBpk(Wv, wv, 1, lane);
    const float bnk = bk[col], bnv = bv[col];
#pragma unroll
    for (int rt = 0; rt < 8; ++rt) {
      v16h a0 = loadA(tok + rt * 16 * 64, lane);
      v16h a1 = loadA(tok + rt * 16 * 64 + 32, lane);
      v8f ck = {}, cv = {};
      ck = wmma16(a0, kb0, ck);
      ck = wmma16(a1, kb1, ck);
      cv = wmma16(a0, vb0, cv);
      cv = wmma16(a1, vb1, cv);
#pragma unroll
      for (int r = 0; r < 8; ++r) {
        int row = rt * 16 + r + 8 * hi;
        khs[row * 64 + col] = (_Float16)(ck[r] + bnk);
        vhs[row * 64 + col] = (_Float16)(cv[r] + bnv);
      }
    }
    v16h qb0 = loadBpk(Wq, wv, 0, lane), qb1 = loadBpk(Wq, wv, 1, lane);
    const float bnq = bq[col];
#pragma unroll
    for (int rt = 0; rt < 2; ++rt) {
      v16h a0 = loadA(qin + rt * 16 * 64, lane);
      v16h a1 = loadA(qin + rt * 16 * 64 + 32, lane);
      v8f c = {};
      c = wmma16(a0, qb0, c);
      c = wmma16(a1, qb1, c);
#pragma unroll
      for (int r = 0; r < 8; ++r)
        qhs[(rt * 16 + r + 8 * hi) * 64 + col] = (_Float16)(c[r] + bnq);
    }
  }
  __syncthreads();

  // ---- 1x4 attention per (edge, head): 128 threads == 32 edges x 4 heads ----
  {
    const int el = tid >> 2, hh = tid & 3;
    float q[16];
    ld16(&qhs[el * 64 + hh * 16], q);
    float sc[4], mx = -1e30f;
#pragma unroll
    for (int t = 0; t < 4; ++t) {
      float kr[16];
      ld16(&khs[(el * 4 + t) * 64 + hh * 16], kr);
      float dot = 0.f;
#pragma unroll
      for (int d = 0; d < 16; ++d) dot += q[d] * kr[d];
      sc[t] = dot * SCL;
      mx = fmaxf(mx, sc[t]);
    }
    float o[16] = {0, 0, 0, 0, 0, 0, 0, 0, 0, 0, 0, 0, 0, 0, 0, 0};
    float l = 0.f;
#pragma unroll
    for (int t = 0; t < 4; ++t) {
      float w = __expf(sc[t] - mx);
      l += w;
      float vr[16];
      ld16(&vhs[(el * 4 + t) * 64 + hh * 16], vr);
#pragma unroll
      for (int d = 0; d < 16; ++d) o[d] += w * vr[d];
    }
    float inv = 1.f / l;
#pragma unroll
    for (int d = 0; d < 16; ++d)
      ohs[el * 64 + hh * 16 + d] = (_Float16)(o[d] * inv);
  }
  __syncthreads();

  // ---- O projection + residual(x_cen) + atomic scatter into out_cen acc ----
  {
    v16h ob0 = loadBpk(Wo, wv, 0, lane), ob1 = loadBpk(Wo, wv, 1, lane);
    const float bn = bo[col];
#pragma unroll
    for (int rt = 0; rt < 2; ++rt) {
      v16h a0 = loadA(ohs + rt * 16 * 64, lane);
      v16h a1 = loadA(ohs + rt * 16 * 64 + 32, lane);
      v8f c = {};
      c = wmma16(a0, ob0, c);
      c = wmma16(a1, ob1, c);
#pragma unroll
      for (int r = 0; r < 8; ++r) {
        int le = rt * 16 + r + 8 * hi;
        int dn = dstI[eb + le];
        float xc = h[(size_t)dn * 64 + col];
        atomicAdd(&acc[(size_t)dn * 64 + col], c[r] + bn + xc);
      }
    }
  }
}

// ---------------- elementwise / scatter kernels -----------------------------------

__global__ void k_zero(float* p, size_t n) {
  size_t i = (size_t)blockIdx.x * blockDim.x + threadIdx.x;
  size_t st = (size_t)gridDim.x * blockDim.x;
  for (; i < n; i += st) p[i] = 0.f;
}

__global__ void k_deg(const int* __restrict__ dstI, float* __restrict__ deg) {
  int i = blockIdx.x * blockDim.x + threadIdx.x;
  if (i < Ee) atomicAdd(&deg[dstI[i]], 1.f);
}

__global__ void k_deg_fin(float* deg) {
  int i = blockIdx.x * blockDim.x + threadIdx.x;
  if (i < Nn) deg[i] = 1.f / fmaxf(deg[i], 1.f);
}

// acc[dst] += xs[src, sp, :]   (one thread per 4 channels, b128 source read)
__global__ void k_scatter(const int* __restrict__ srcI, const int* __restrict__ dstI,
                          const float* __restrict__ xs, int sp,
                          float* __restrict__ acc) {
  int i = blockIdx.x * blockDim.x + threadIdx.x;
  if (i >= Ee * 16) return;
  int e = i >> 4, c4 = (i & 15) << 2;
  v4f v = *(const v4f*)&xs[((size_t)srcI[e] * Ss + sp) * 64 + c4];
  float* dp = &acc[(size_t)dstI[e] * 64 + c4];
  atomicAdd(dp + 0, v[0]);
  atomicAdd(dp + 1, v[1]);
  atomicAdd(dp + 2, v[2]);
  atomicAdd(dp + 3, v[3]);
}

// xs[n, s, :] = acc[n, :] * deg_inv[n]
__global__ void k_hop_scale(const float* __restrict__ acc,
                            const float* __restrict__ dinv,
                            float* __restrict__ xs, int s) {
  int i = blockIdx.x * blockDim.x + threadIdx.x;
  if (i >= Nn * 64) return;
  int n = i >> 6, c = i & 63;
  xs[((size_t)n * Ss + s) * 64 + c] = acc[i] * dinv[n];
}

__global__ void k_ln(const float* __restrict__ x, const float* __restrict__ s,
                     const float* __restrict__ b, _Float16* __restrict__ y, int M) {
  int i = blockIdx.x * blockDim.x + threadIdx.x;
  if (i >= M) return;
  const float* r = x + (size_t)i * 64;
  float sum = 0.f, sq = 0.f;
#pragma unroll
  for (int c = 0; c < 64; ++c) { float v = r[c]; sum += v; sq += v * v; }
  float mean = sum * (1.f / 64.f);
  float var  = sq * (1.f / 64.f) - mean * mean;
  float rstd = rsqrtf(var + LN_EPS);
  _Float16* o = y + (size_t)i * 64;
#pragma unroll
  for (int c = 0; c < 64; ++c)
    o[c] = (_Float16)((r[c] - mean) * rstd * s[c] + b[c]);
}

// global branch: per (node-token, head) online softmax over 128 codebook keys
__global__ void k_gattn(const _Float16* __restrict__ gq, const float* __restrict__ gkp,
                        const float* __restrict__ gvp, const float* __restrict__ xs,
                        float* __restrict__ og) {
  int i = blockIdx.x * blockDim.x + threadIdx.x;
  if (i >= Nn * Ss * Hh) return;
  int hh = i & 3;
  size_t ns = (size_t)(i >> 2);
  size_t base = ns * 64 + (size_t)hh * 16;
  float q[16];
  ld16(gq + base, q);
  float m = -1e30f, l = 0.f;
  float acc[16] = {0, 0, 0, 0, 0, 0, 0, 0, 0, 0, 0, 0, 0, 0, 0, 0};
  for (int k = 0; k < Kk; ++k) {
    const float* kr = gkp + k * 64 + hh * 16;
    float dot = 0.f;
#pragma unroll
    for (int d = 0; d < 16; ++d) dot += q[d] * kr[d];
    dot *= SCL;
    float mn = fmaxf(m, dot);
    float corr = __expf(m - mn);
    float w = __expf(dot - mn);
    l = l * corr + w;
    const float* vr = gvp + k * 64 + hh * 16;
#pragma unroll
    for (int d = 0; d < 16; ++d) acc[d] = acc[d] * corr + w * vr[d];
    m = mn;
  }
  float inv = 1.f / l;
#pragma unroll
  for (int d = 0; d < 16; ++d) og[base + d] = acc[d] * inv + xs[base + d];
}

// local branch: 3-token attention per (node, token, head)
__global__ void k_lattn(const _Float16* __restrict__ q, const _Float16* __restrict__ k,
                        const _Float16* __restrict__ v, _Float16* __restrict__ o) {
  int i = blockIdx.x * blockDim.x + threadIdx.x;
  if (i >= Nn * Ss * Hh) return;
  int hh = i & 3;
  int ns = i >> 2;
  int s  = ns % 3;
  int n3 = ns - s;                 // first token row of this node
  float qv[16];
  ld16(q + (size_t)ns * 64 + hh * 16, qv);
  float sc[3], mx = -1e30f;
#pragma unroll
  for (int t = 0; t < 3; ++t) {
    float kr[16];
    ld16(k + ((size_t)(n3 + t)) * 64 + hh * 16, kr);
    float dot = 0.f;
#pragma unroll
    for (int d = 0; d < 16; ++d) dot += qv[d] * kr[d];
    sc[t] = dot * SCL;
    mx = fmaxf(mx, sc[t]);
  }
  float l = 0.f;
  float out[16] = {0, 0, 0, 0, 0, 0, 0, 0, 0, 0, 0, 0, 0, 0, 0, 0};
#pragma unroll
  for (int t = 0; t < 3; ++t) {
    float w = __expf(sc[t] - mx);
    l += w;
    float vr[16];
    ld16(v + ((size_t)(n3 + t)) * 64 + hh * 16, vr);
#pragma unroll
    for (int d = 0; d < 16; ++d) out[d] += w * vr[d];
  }
  float inv = 1.f / l;
#pragma unroll
  for (int d = 0; d < 16; ++d)
    o[(size_t)ns * 64 + hh * 16 + d] = (_Float16)(out[d] * inv);
}

// convert 14 fp32 64x64 weights to f16 in WMMA B-fragment order:
// packed[mat][ct][kc][lane][e] = W[((e&7) + 8*(lane>>4) + 16*(e>>3) + 32*kc)*64 + ct*16 + (lane&15)]
struct WSrc { const float* p[14]; };
__global__ void k_cvt_w(WSrc w, _Float16* out) {
  int i = blockIdx.x * blockDim.x + threadIdx.x;
  if (i >= 14 * 4096) return;
  int mat  = i >> 12;
  int r    = i & 4095;
  int e    = r & 15;
  int lane = (r >> 4) & 31;
  int kc   = (r >> 9) & 1;
  int ct   = (r >> 10) & 3;
  int n = lane & 15, hi = lane >> 4;
  int k = (e & 7) + 8 * hi + 16 * (e >> 3) + 32 * kc;
  out[i] = (_Float16)w.p[mat][k * 64 + ct * 16 + n];
}

// ---------------- host launcher ---------------------------------------------------

extern "C" void kernel_launch(void* const* d_in, const int* in_sizes, int n_in,
                              void* d_out, int out_size, void* d_ws, size_t ws_size,
                              hipStream_t stream) {
  (void)in_sizes; (void)n_in; (void)out_size; (void)ws_size;

  const float* x        = (const float*)d_in[0];
  const int*   ei       = (const int*)d_in[1];
  const int*   srcI     = ei;
  const int*   dstI     = ei + Ee;
  const float* b_in     = (const float*)d_in[3];
  const float* nti_ln_s = (const float*)d_in[4];
  const float* nti_ln_b = (const float*)d_in[5];
  const float* nti_bq   = (const float*)d_in[7];
  const float* nti_bk   = (const float*)d_in[9];
  const float* nti_bv   = (const float*)d_in[11];
  const float* nti_bo   = (const float*)d_in[13];
  const float* tc_ln1_s = (const float*)d_in[14];
  const float* tc_ln1_b = (const float*)d_in[15];
  const float* tc_bq    = (const float*)d_in[17];
  const float* tc_bk    = (const float*)d_in[19];
  const float* tc_bv    = (const float*)d_in[21];
  const float* tc_bo    = (const float*)d_in[23];
  const float* tc_ln2_s = (const float*)d_in[24];
  const float* tc_ln2_b = (const float*)d_in[25];
  const float* ffn_b1   = (const float*)d_in[27];
  const float* ffn_b2   = (const float*)d_in[29];
  const float* g_bq     = (const float*)d_in[31];
  const float* g_bk     = (const float*)d_in[33];
  const float* g_bv     = (const float*)d_in[35];
  const float* codebook = (const float*)d_in[36];
  const float* hopwise  = (const float*)d_in[37];

  // workspace carve-up
  char* base = (char*)d_ws;
  size_t off = 0;
  auto take = [&](size_t bytes) -> void* {
    off = (off + 255) & ~(size_t)255;
    void* r = base + off;
    off += bytes;
    return r;
  };
  _Float16* wf16 = (_Float16*)take((size_t)14 * 4096 * sizeof(_Float16));
  float* gkp     = (float*)take((size_t)Kk * 64 * 4);
  float* gvp     = (float*)take((size_t)Kk * 64 * 4);
  float* dinv    = (float*)take((size_t)Nn * 4);
  float* h       = (float*)take((size_t)Nn * 64 * 4);
  float* acc     = (float*)take((size_t)Nn * 64 * 4);
  float* xs      = (float*)take((size_t)Nn * Ss * 64 * 4);
  float* xl      = (float*)take((size_t)Nn * Ss * 64 * 4);
  float* og      = (float*)take((size_t)Nn * Ss * 64 * 4);
  _Float16* yh   = (_Float16*)take((size_t)Nn * Ss * 64 * 2);
  _Float16* qh   = (_Float16*)take((size_t)Nn * Ss * 64 * 2);
  _Float16* kh   = (_Float16*)take((size_t)Nn * Ss * 64 * 2);
  _Float16* vh   = (_Float16*)take((size_t)Nn * Ss * 64 * 2);

  // f16 weight slots: 0 W_in, 1..4 nti q/k/v/o, 5..8 tc q/k/v/o,
  //                   9 ffn_W1, 10 ffn_W2, 11 g_Wq, 12 g_Wk, 13 g_Wv
  WSrc wsrc;
  const int widx[14] = {2, 6, 8, 10, 12, 16, 18, 20, 22, 26, 28, 30, 32, 34};
  for (int i = 0; i < 14; ++i) wsrc.p[i] = (const float*)d_in[widx[i]];
  k_cvt_w<<<(14 * 4096) / 256, 256, 0, stream>>>(wsrc, wf16);

  const int MS = Nn * Ss;                       // 150000 rows, 9375 tiles

  // degree (clamped) -> inverse
  k_zero<<<1024, 256, 0, stream>>>(dinv, (size_t)Nn);
  k_deg<<<(Ee + 255) / 256, 256, 0, stream>>>(dstI, dinv);
  k_deg_fin<<<(Nn + 255) / 256, 256, 0, stream>>>(dinv);

  // h = x @ W_in + b_in   (3125 tiles, MT=5)
  k_gemm64<float, 0, 5><<<Nn / 80, 128, 0, stream>>>(x, wf16 + 0 * 4096, b_in, h,
                                                     nullptr, nullptr, nullptr);

  // fused edge block -> acc (out_cen numerator)
  k_zero<<<2048, 256, 0, stream>>>(acc, (size_t)Nn * 64);
  k_edge<<<Ee / 32, 128, 0, stream>>>(h, srcI, dstI, nti_ln_s, nti_ln_b,
                                      wf16 + 1 * 4096, nti_bq, wf16 + 2 * 4096, nti_bk,
                                      wf16 + 3 * 4096, nti_bv, wf16 + 4 * 4096, nti_bo,
                                      acc);
  k_hop_scale<<<(Nn * 64) / 256, 256, 0, stream>>>(acc, dinv, xs, 0);

  // 2 hops of mean message passing
  for (int hop = 0; hop < 2; ++hop) {
    k_zero<<<2048, 256, 0, stream>>>(acc, (size_t)Nn * 64);
    k_scatter<<<(Ee * 16) / 256, 256, 0, stream>>>(srcI, dstI, xs, hop, acc);
    k_hop_scale<<<(Nn * 64) / 256, 256, 0, stream>>>(acc, dinv, xs, hop + 1);
  }

  // global branch: project codebook K/V (8 tiles, MT=2), queries, attention
  k_gemm64<float, 0, 2><<<Kk / 32, 128, 0, stream>>>(codebook, wf16 + 12 * 4096, g_bk,
                                                     gkp, nullptr, nullptr, nullptr);
  k_gemm64<float, 0, 2><<<Kk / 32, 128, 0, stream>>>(codebook, wf16 + 13 * 4096, g_bv,
                                                     gvp, nullptr, nullptr, nullptr);
  k_gemm64<float, 1, 5><<<MS / 80, 128, 0, stream>>>(xs, wf16 + 11 * 4096, g_bq, qh,
                                                     nullptr, nullptr, nullptr);
  k_gattn<<<(Nn * Ss * Hh + 255) / 256, 256, 0, stream>>>(qh, gkp, gvp, xs, og);

  // local branch
  k_ln<<<(MS + 255) / 256, 256, 0, stream>>>(xs, tc_ln1_s, tc_ln1_b, yh, MS);
  k_gemm64<_Float16, 1, 5><<<MS / 80, 128, 0, stream>>>(yh, wf16 + 5 * 4096, tc_bq, qh,
                                                        nullptr, nullptr, nullptr);
  k_gemm64<_Float16, 1, 5><<<MS / 80, 128, 0, stream>>>(yh, wf16 + 6 * 4096, tc_bk, kh,
                                                        nullptr, nullptr, nullptr);
  k_gemm64<_Float16, 1, 5><<<MS / 80, 128, 0, stream>>>(yh, wf16 + 7 * 4096, tc_bv, vh,
                                                        nullptr, nullptr, nullptr);
  k_lattn<<<(Nn * Ss * Hh + 255) / 256, 256, 0, stream>>>(qh, kh, vh, yh);
  // xl = xs + o @ Wo + bo
  k_gemm64<_Float16, 3, 5><<<MS / 80, 128, 0, stream>>>(yh, wf16 + 8 * 4096, tc_bo, xl,
                                                        xs, nullptr, nullptr);
  // FFN + final combine into d_out
  k_ln<<<(MS + 255) / 256, 256, 0, stream>>>(xl, tc_ln2_s, tc_ln2_b, yh, MS);
  k_gemm64<_Float16, 2, 5><<<MS / 80, 128, 0, stream>>>(yh, wf16 + 9 * 4096, ffn_b1, qh,
                                                        nullptr, nullptr, nullptr);
  k_gemm64<_Float16, 4, 5><<<MS / 80, 128, 0, stream>>>(qh, wf16 + 10 * 4096, ffn_b2,
                                                        d_out, xl, og, hopwise);
}
